// MyLSTM1_44581760532733
// MI455X (gfx1250) — matrix operand order repaired and checked by reading, not compiled, not run on hardware
//
#include <hip/hip_runtime.h>
#include <hip/hip_bf16.h>

// ---------------------------------------------------------------------------
// LSTM on gfx1250 (MI455X): f16 WMMA (f32 accumulate) for both the big
// input-projection GEMM and the 256 recurrent GEMMs. X, h and the weights are
// pre-swizzled into the CDNA5 wave32 WMMA fragment layouts so every fragment
// load is one contiguous 32B-per-lane global load; the GEMM inner loops are
// software-pipelined so fragment loads overlap WMMA execution.
//
// Workspace layout (bytes), total 677,388,288 (~646 MiB):
//   Xp      f16  [T=256][4096 tiles][256]         = 536,870,912
//   Xs      f16  [131072x512] A-fragment order    = 134,217,728
//   Wx_swz  f16  [2048x512]   B-fragment order    =   2,097,152
//   Wh_swz  f16  [2048x512]   B-fragment order    =   2,097,152
//   bsum    fp32 [2048]                           =       8,192
//   c       fp32 [512x512]                        =   1,048,576
//   h_swz0  f16  [512x512]    A-fragment order    =     524,288
//   h_swz1  f16  [512x512]    A-fragment order    =     524,288
// ---------------------------------------------------------------------------

typedef __attribute__((ext_vector_type(16))) _Float16 v16h;
typedef __attribute__((ext_vector_type(8)))  _Float16 v8h;
typedef __attribute__((ext_vector_type(8)))  float    v8f;
typedef __attribute__((ext_vector_type(4)))  float    v4f;

#define LSTM_T   256
#define LSTM_B   512
#define LSTM_H   512
#define LSTM_IN  512
#define G4H      2048   // 4*H

// ---- fragment layouts (CDNA5 ISA, wave32) -----------------------
// A (16x32 f16), lane l, half j:  m = l&15,  k = (j>>3)*16 + (l>>4)*8 + (j&7)
// B (32x16 f16), lane l, half j:  n = l&15,  k = (j>>3)*16 + (l>>4)*8 + (j&7)
// C (16x16 f32), lane l, vgpr r:  m = (l>>4)*8 + r, n = l&15

// Build fused+swizzled f16 weight matrix [2048 x 512] (gate order i,f,g,o)
// in B-fragment order: offset = ((nt*16 + kt)*512 + l*16 + j) halves.
__global__ void cvt_weights_kernel(const float* __restrict__ Wi,
                                   const float* __restrict__ Wf,
                                   const float* __restrict__ Wg,
                                   const float* __restrict__ Wo,
                                   _Float16* __restrict__ dst) {
    int base = (blockIdx.x * 256 + threadIdx.x) * 4;   // 1024 blocks * 256 thr * 4
    #pragma unroll
    for (int u = 0; u < 4; ++u) {
        int o      = base + u;
        int tile   = o >> 9;           // (nt*16 + kt)
        int within = o & 511;
        int l      = within >> 4;
        int j      = within & 15;
        int nt     = tile >> 4;
        int kt     = tile & 15;
        int n      = nt * 16 + (l & 15);          // 0..2047 fused row
        int k      = kt * 32 + (j >> 3) * 16 + (l >> 4) * 8 + (j & 7); // 0..511
        int g      = n >> 9;
        int row    = n & 511;
        const float* W = (g == 0) ? Wi : (g == 1) ? Wf : (g == 2) ? Wg : Wo;
        dst[o] = (_Float16)W[(size_t)row * 512 + k];
    }
}

// Pre-swizzle X [131072 x 512] fp32 into A-fragment f16 layout.
// One thread = one lane-fragment (16 halves): frag = (gm*16 + kt)*32 + l.
__global__ void cvt_x_kernel(const float* __restrict__ X,
                             _Float16* __restrict__ Xs) {
    int frag = blockIdx.x * 256 + threadIdx.x;   // 16384 blocks -> 4,194,304 frags
    int l    = frag & 31;
    int kt   = (frag >> 5) & 15;
    int gm   = frag >> 9;
    int lh   = l >> 4;
    int row  = gm * 16 + (l & 15);
    const float* src = X + (size_t)row * LSTM_IN + kt * 32 + lh * 8;
    v4f a0 = *(const v4f*)(src);
    v4f a1 = *(const v4f*)(src + 4);
    v4f a2 = *(const v4f*)(src + 16);
    v4f a3 = *(const v4f*)(src + 20);
    v16h o;
    #pragma unroll
    for (int j = 0; j < 4; ++j) {
        o[j]      = (_Float16)a0[j];
        o[j + 4]  = (_Float16)a1[j];
        o[j + 8]  = (_Float16)a2[j];
        o[j + 12] = (_Float16)a3[j];
    }
    *(v16h*)(Xs + (size_t)frag * 16) = o;
}

// Zero c state and h_swz0; build bsum = bx + bh (fused gate order).
__global__ void init_state_kernel(const float* __restrict__ bxi, const float* __restrict__ bxf,
                                  const float* __restrict__ bxg, const float* __restrict__ bxo,
                                  const float* __restrict__ bhi, const float* __restrict__ bhf,
                                  const float* __restrict__ bhg, const float* __restrict__ bho,
                                  float* __restrict__ bsum, float* __restrict__ c,
                                  _Float16* __restrict__ h0) {
    int i = blockIdx.x * 256 + threadIdx.x;   // 1024 blocks * 256 = 262144 threads
    c[i]  = 0.0f;
    h0[i] = (_Float16)0.0f;
    if (i < G4H) {
        int g = i >> 9, r = i & 511;
        const float* bx = (g == 0) ? bxi : (g == 1) ? bxf : (g == 2) ? bxg : bxo;
        const float* bh = (g == 0) ? bhi : (g == 1) ? bhf : (g == 2) ? bhg : bho;
        bsum[i] = bx[r] + bh[r];
    }
}

// Xp = X @ Wx^T (bias added later). Each wave: one M-tile (16 rows of T*B)
// x 8 N-tiles, K-loop 16. All 8 B fragments loaded per K-step before the
// WMMA block so the 9 loads of step kt overlap the WMMAs of step kt-1.
// Output stored f16 in C-fragment-tile order: ((gm*128 + nt)*256 + lane*8 + r).
__global__ void xp_gemm_kernel(const _Float16* __restrict__ Xs,
                               const _Float16* __restrict__ Wx,
                               _Float16* __restrict__ Xp) {
    int wave = (blockIdx.x * blockDim.x + threadIdx.x) >> 5;
    int lane = threadIdx.x & 31;
    int gm   = wave >> 4;        // M tile index in [0, 8192)
    int ntg  = wave & 15;        // covers N-tiles ntg*8 .. ntg*8+7

    const _Float16* Ab = Xs + (size_t)gm * 16 * 512 + lane * 16;
    const _Float16* Bb = Wx + (size_t)(ntg * 8) * 16 * 512 + lane * 16;

    v8f acc[8];
    #pragma unroll
    for (int i = 0; i < 8; ++i) acc[i] = (v8f){0.f,0.f,0.f,0.f,0.f,0.f,0.f,0.f};

    for (int kt = 0; kt < 16; ++kt) {
        v16h af = *(const v16h*)(Ab + kt * 512);
        v16h bf[8];
        #pragma unroll
        for (int i = 0; i < 8; ++i)
            bf[i] = *(const v16h*)(Bb + (size_t)i * 16 * 512 + kt * 512);
        #pragma unroll
        for (int i = 0; i < 8; ++i)
            acc[i] = __builtin_amdgcn_wmma_f32_16x16x32_f16(
                false, af, false, bf[i], (short)0, acc[i], false, false);
    }
    #pragma unroll
    for (int i = 0; i < 8; ++i) {
        int nt = ntg * 8 + i;
        v8h st;
        #pragma unroll
        for (int r = 0; r < 8; ++r) st[r] = (_Float16)acc[i][r];
        *(v8h*)(Xp + (size_t)(gm * 128 + nt) * 256 + lane * 8) = st;
    }
}

__device__ __forceinline__ float sigmoid_f(float x) {
    return 1.0f / (1.0f + __expf(-x));
}
__device__ __forceinline__ float tanh_f(float x) {
    x = fminf(15.0f, fmaxf(-15.0f, x));
    float e = __expf(2.0f * x);
    return (e - 1.0f) / (e + 1.0f);
}

// One recurrence step. Grid = 512 blocks (32 M-tiles x 16 col-blocks of 32),
// 256 threads (8 waves). Wave w = (gate, sub) computes z-tile
// (mt, nt = gate*32 + cb*2 + sub) = h_swz(A) x Wh_swz(B) + Xp + bsum, with a
// depth-4 software pipeline on the K loop. Phase 2 combines the 4 gates via
// LDS, updates c, writes h and re-swizzles h for the next step (ping-pong).
__global__ void lstm_step_kernel(const _Float16* __restrict__ Xp_t,
                                 const _Float16* __restrict__ Wh,
                                 const float* __restrict__ bsum,
                                 const _Float16* __restrict__ h_src,
                                 _Float16* __restrict__ h_dst,
                                 float* __restrict__ cstate,
                                 float* __restrict__ out_t) {
    __shared__ float zsh[8 * 256];   // 8 z-tiles of 16x16 fp32

    int mt   = blockIdx.x >> 4;      // 0..31 batch tile
    int cb   = blockIdx.x & 15;      // 0..15 hidden col block (32 cols)
    int w    = threadIdx.x >> 5;     // wave 0..7
    int lane = threadIdx.x & 31;
    int gate = w >> 1;
    int sub  = w & 1;
    int nt   = gate * 32 + cb * 2 + sub;

    // Accumulator starts from Xp + fused bias.
    v8h xp = *(const v8h*)(Xp_t + (size_t)(mt * 128 + nt) * 256 + lane * 8);
    float bs = bsum[nt * 16 + (lane & 15)];
    v8f acc;
    #pragma unroll
    for (int r = 0; r < 8; ++r) acc[r] = (float)xp[r] + bs;

    const _Float16* Ab = h_src + (size_t)mt * 16 * 512 + lane * 16;
    const _Float16* Bb = Wh    + (size_t)nt * 16 * 512 + lane * 16;

    // Depth-4 pipelined K loop: 16 x v_wmma_f32_16x16x32_f16.
    v16h Af[4], Bf[4];
    #pragma unroll
    for (int p = 0; p < 4; ++p) {
        Af[p] = *(const v16h*)(Ab + p * 512);
        Bf[p] = *(const v16h*)(Bb + p * 512);
    }
    #pragma unroll
    for (int kt = 0; kt < 16; ++kt) {
        v16h a = Af[kt & 3];
        v16h b = Bf[kt & 3];
        if (kt + 4 < 16) {
            Af[kt & 3] = *(const v16h*)(Ab + (kt + 4) * 512);
            Bf[kt & 3] = *(const v16h*)(Bb + (kt + 4) * 512);
        }
        acc = __builtin_amdgcn_wmma_f32_16x16x32_f16(
            false, a, false, b, (short)0, acc, false, false);
    }

    #pragma unroll
    for (int r = 0; r < 8; ++r)
        zsh[w * 256 + lane * 8 + r] = acc[r];

    __syncthreads();

    // Phase 2: 16 rows x 32 hidden cols = 512 elements, 2 per thread.
    for (int e = threadIdx.x; e < 512; e += 256) {
        int mr = e >> 5;              // 0..15 row within tile
        int nc = e & 31;              // 0..31 hidden col within block
        int r     = mr & 7;
        int lbase = ((mr >> 3) * 16 + (nc & 15)) * 8 + r;
        int sub2  = (nc >> 4);
        float zi = zsh[(0 * 2 + sub2) * 256 + lbase];
        float zf = zsh[(1 * 2 + sub2) * 256 + lbase];
        float zg = zsh[(2 * 2 + sub2) * 256 + lbase];
        float zo = zsh[(3 * 2 + sub2) * 256 + lbase];

        float ig = sigmoid_f(zi);
        float fg = sigmoid_f(zf);
        float gg = tanh_f(zg);
        float og = sigmoid_f(zo);

        int m = mt * 16 + mr;
        int n = cb * 32 + nc;
        size_t ci = (size_t)m * LSTM_H + n;
        float cn = fg * cstate[ci] + ig * gg;
        cstate[ci] = cn;
        float h = og * tanh_f(cn);
        out_t[ci] = h;

        // Re-swizzle h into A-fragment order for next step (A tile (mt, cb)).
        int l2 = ((nc >> 3) & 1) * 16 + (m & 15);
        int j2 = ((nc >> 4) << 3) | (nc & 7);
        h_dst[(size_t)(mt * 16 + cb) * 512 + l2 * 16 + j2] = (_Float16)h;
    }
}

extern "C" void kernel_launch(void* const* d_in, const int* in_sizes, int n_in,
                              void* d_out, int out_size, void* d_ws, size_t ws_size,
                              hipStream_t stream) {
    (void)in_sizes; (void)n_in; (void)out_size; (void)ws_size;
    const float* X    = (const float*)d_in[0];
    const float* W_ii = (const float*)d_in[1];
    const float* b_ii = (const float*)d_in[2];
    const float* W_ih = (const float*)d_in[3];
    const float* b_ih = (const float*)d_in[4];
    const float* W_fi = (const float*)d_in[5];
    const float* b_fi = (const float*)d_in[6];
    const float* W_fh = (const float*)d_in[7];
    const float* b_fh = (const float*)d_in[8];
    const float* W_gi = (const float*)d_in[9];
    const float* b_gi = (const float*)d_in[10];
    const float* W_gh = (const float*)d_in[11];
    const float* b_gh = (const float*)d_in[12];
    const float* W_oi = (const float*)d_in[13];
    const float* b_oi = (const float*)d_in[14];
    const float* W_oh = (const float*)d_in[15];
    const float* b_oh = (const float*)d_in[16];

    char* ws = (char*)d_ws;
    _Float16* Xp     = (_Float16*)ws;                        // 536,870,912 B
    _Float16* Xs     = (_Float16*)(ws + 536870912ull);       // 134,217,728 B
    _Float16* Wx_swz = (_Float16*)(ws + 671088640ull);       //   2,097,152 B
    _Float16* Wh_swz = (_Float16*)(ws + 673185792ull);       //   2,097,152 B
    float*    bsum   = (float*)   (ws + 675282944ull);       //       8,192 B
    float*    cst    = (float*)   (ws + 675291136ull);       //   1,048,576 B
    _Float16* h0     = (_Float16*)(ws + 676339712ull);       //     524,288 B
    _Float16* h1     = (_Float16*)(ws + 676864000ull);       //     524,288 B

    cvt_weights_kernel<<<1024, 256, 0, stream>>>(W_ii, W_fi, W_gi, W_oi, Wx_swz);
    cvt_weights_kernel<<<1024, 256, 0, stream>>>(W_ih, W_fh, W_gh, W_oh, Wh_swz);
    cvt_x_kernel<<<16384, 256, 0, stream>>>(X, Xs);
    init_state_kernel<<<1024, 256, 0, stream>>>(b_ii, b_fi, b_gi, b_oi,
                                                b_ih, b_fh, b_gh, b_oh,
                                                bsum, cst, h0);
    // 131072 waves = 16384 blocks of 8 waves
    xp_gemm_kernel<<<16384, 256, 0, stream>>>(Xs, Wx_swz, Xp);

    float* out = (float*)d_out;
    for (int t = 0; t < LSTM_T; ++t) {
        const _Float16* hsrc = (t & 1) ? h1 : h0;
        _Float16*       hdst = (t & 1) ? h0 : h1;
        lstm_step_kernel<<<512, 256, 0, stream>>>(
            Xp + (size_t)t * 4096 * 256, Wh_swz, bsum,
            hsrc, hdst, cst,
            out + (size_t)t * LSTM_B * LSTM_H);
    }
}
